// TransformerBlock_51110110823045
// MI455X (gfx1250) — matrix operand order repaired and checked by reading, not compiled, not run on hardware
//
#include <hip/hip_runtime.h>
#include <hip/hip_bf16.h>
#include <math.h>
#include <stdint.h>

// ---------------------------------------------------------------------------
// Types for CDNA5 WMMA (wave32): v_wmma_f32_16x16x32_f16
// ---------------------------------------------------------------------------
typedef _Float16 v8h  __attribute__((ext_vector_type(8)));
typedef _Float16 v16h __attribute__((ext_vector_type(16)));
typedef float    v8f  __attribute__((ext_vector_type(8)));

static __device__ __forceinline__ v16h concat8(v8h lo, v8h hi) {
    return __builtin_shufflevector(lo, hi, 0,1,2,3,4,5,6,7,8,9,10,11,12,13,14,15);
}

// ---------------------------------------------------------------------------
// CDNA5 async global -> LDS copy (16 bytes per lane), tracked by ASYNCcnt.
// LDS operand is the wave-relative LDS byte offset = low 32 bits of the
// generic pointer (addrspacecast AS3->generic is {aperture_hi, lds_offset}).
// ---------------------------------------------------------------------------
static __device__ __forceinline__ void async_copy16(const _Float16* gp, _Float16* lp) {
#if defined(__HIP_DEVICE_COMPILE__)
    const unsigned l = (unsigned)(uintptr_t)lp;
    asm volatile("global_load_async_to_lds_b128 %0, %1, off" :: "v"(l), "v"(gp) : "memory");
#else
    *(v8h*)lp = *(const v8h*)gp;
#endif
}
static __device__ __forceinline__ void wait_async() {
#if defined(__HIP_DEVICE_COMPILE__)
    asm volatile("s_wait_asynccnt 0x0" ::: "memory");
#endif
}

// ---------------------------------------------------------------------------
// Batched NT GEMM:  C[M,N] = alpha * A[M,K] * B[N,K]^T  (+ fused epilogue)
// A, B: f16; acc: f32. Block tile 128x128, K-chunk 32, double-buffered LDS
// filled with async global->LDS copies (overlaps the next tile's copies with
// the current tile's 8 v_wmma_f32_16x16x32_f16 per wave).
// 256 threads = 8 wave32; each wave owns a 32x64 sub-tile (2x4 WMMA tiles).
//
// EPI: 0 = store f32
//      1 = store f16
//      2 = store f32 + residual  (Cf[idx] = acc + resid[idx]; in-place OK)
//      3 = relu -> f16
//      4 = multiply by auxh[idx] -> f16   (gated FFN)
// Batch: z -> (b = z / Hdim, h = z % Hdim), element offsets b*s?b + h*s?h.
// ---------------------------------------------------------------------------
template <int EPI>
__global__ void __launch_bounds__(256)
wmma_gemm_nt(const _Float16* __restrict__ A, const _Float16* __restrict__ Bm,
             float* __restrict__ Cf, _Float16* __restrict__ Ch,
             const float* __restrict__ resid, const _Float16* __restrict__ auxh,
             int M, int N, int K, int lda, int ldb, int ldc,
             long long sAb, long long sAh, long long sBb, long long sBh,
             long long sCb, long long sCh, int Hdim, float alpha) {
    __shared__ _Float16 ldsA[2][128][40];  // 40-halfs pitch: 16B aligned, conflict-padded
    __shared__ _Float16 ldsB[2][128][40];

    const int tid  = threadIdx.x;
    const int lane = tid & 31;
    const int wave = tid >> 5;
    const int wr   = wave >> 1;          // 0..3 -> M offset wr*32
    const int wc   = wave & 1;           // 0..1 -> N offset wc*64
    const int l16  = lane & 15;
    const int g    = lane >> 4;          // lane half-group
    const int kqA  = g * 8;              // A frag K pattern: [g*8..+7] and [16+g*8..+7]
    const int kqB  = g * 16;             // unused: B frag uses A's K pattern (kqA)

    const int z = blockIdx.z;
    const int b = z / Hdim;
    const int h = z % Hdim;
    const _Float16* Ab   = A  + (size_t)(b * sAb + h * sAh);
    const _Float16* Bb   = Bm + (size_t)(b * sBb + h * sBh);
    const long long Coff = b * sCb + h * sCh;

    const int row0A = blockIdx.y * 128;  // M rows
    const int row0B = blockIdx.x * 128;  // N rows (rows of B = columns of B^T)
    if (row0A >= M || row0B >= N) return;

    // Staging map: 512 16-byte segments per tile; 2 per thread per matrix.
    const int sr = tid >> 2;             // rows 0..63 (+64 for segment 1)
    const int sc = (tid & 3) << 3;       // 8-half column group

    auto stage = [&](int buf, int kb) {
#pragma unroll
        for (int s = 0; s < 2; ++s) {
            const int r = sr + s * 64;
            async_copy16(Ab + (size_t)(row0A + r) * lda + kb + sc, &ldsA[buf][r][sc]);
            async_copy16(Bb + (size_t)(row0B + r) * ldb + kb + sc, &ldsB[buf][r][sc]);
        }
    };

    v8f acc[2][4];
#pragma unroll
    for (int mi = 0; mi < 2; ++mi)
#pragma unroll
        for (int ni = 0; ni < 4; ++ni) acc[mi][ni] = (v8f)0.0f;

    stage(0, 0);                          // prologue: tile 0 in flight
    int cur = 0;
    for (int kb = 0; kb < K; kb += 32) {
        wait_async();                     // my copies for buf `cur` are done
        __syncthreads();                  // everyone's copies done; prev compute done
        if (kb + 32 < K) stage(cur ^ 1, kb + 32);   // overlap next tile with compute

        v16h af[2];
#pragma unroll
        for (int mi = 0; mi < 2; ++mi) {
            const _Float16* p = &ldsA[cur][wr * 32 + mi * 16 + l16][0];
            af[mi] = concat8(*(const v8h*)(p + kqA), *(const v8h*)(p + kqA + 16));
        }
        v16h bf[4];
#pragma unroll
        for (int ni = 0; ni < 4; ++ni) {
            const _Float16* p = &ldsB[cur][wc * 64 + ni * 16 + l16][0];
            bf[ni] = concat8(*(const v8h*)(p + kqA), *(const v8h*)(p + kqA + 16));
        }
#pragma unroll
        for (int mi = 0; mi < 2; ++mi)
#pragma unroll
            for (int ni = 0; ni < 4; ++ni)
                acc[mi][ni] = __builtin_amdgcn_wmma_f32_16x16x32_f16(
                    false, af[mi], false, bf[ni], (short)0, acc[mi][ni], false, false);
        cur ^= 1;
    }

    // Epilogue: C tile layout — VGPR r holds row (lane<16 ? r : 8+r), col = lane%16.
#pragma unroll
    for (int mi = 0; mi < 2; ++mi) {
#pragma unroll
        for (int ni = 0; ni < 4; ++ni) {
            const int mbase = row0A + wr * 32 + mi * 16 + ((lane < 16) ? 0 : 8);
            const int n     = row0B + wc * 64 + ni * 16 + l16;
#pragma unroll
            for (int r = 0; r < 8; ++r) {
                const long long idx = Coff + (long long)(mbase + r) * ldc + n;
                const float val = acc[mi][ni][r] * alpha;
                if      (EPI == 0) Cf[idx] = val;
                else if (EPI == 1) Ch[idx] = (_Float16)val;
                else if (EPI == 2) Cf[idx] = val + resid[idx];
                else if (EPI == 3) Ch[idx] = (_Float16)fmaxf(val, 0.0f);
                else               Ch[idx] = (_Float16)(val * (float)auxh[idx]);
            }
        }
    }
}

// ---------------------------------------------------------------------------
// RMSNorm (row of D=2048, 256 threads x 8 elems) + cast to f16
// ---------------------------------------------------------------------------
__global__ void __launch_bounds__(256)
rmsnorm_cast_kernel(const float* __restrict__ x, const float* __restrict__ gw,
                    _Float16* __restrict__ out, int D) {
    __shared__ float red[256];
    const int tid = threadIdx.x;
    const size_t base = (size_t)blockIdx.x * D;
    float v[8];
    float local = 0.0f;
#pragma unroll
    for (int i = 0; i < 8; ++i) {
        v[i] = x[base + tid + i * 256];
        local += v[i] * v[i];
    }
    red[tid] = local;
    __syncthreads();
    for (int off = 128; off > 0; off >>= 1) {
        if (tid < off) red[tid] += red[tid + off];
        __syncthreads();
    }
    const float norm = rsqrtf(red[0] / (float)D + 1e-6f);
#pragma unroll
    for (int i = 0; i < 8; ++i) {
        const int c = tid + i * 256;
        out[base + c] = (_Float16)(v[i] * norm * gw[c]);
    }
}

// ---------------------------------------------------------------------------
// Row softmax in place (row length 2048) + f16 copy of the probabilities so
// the attention GEMM can use the async f16 staging path.
// ---------------------------------------------------------------------------
__global__ void __launch_bounds__(256)
softmax_rows_kernel(float* __restrict__ s, _Float16* __restrict__ p16, int n) {
    __shared__ float red[256];
    const int tid = threadIdx.x;
    const size_t base = (size_t)blockIdx.x * n;
    float v[8];
    float mx = -INFINITY;
#pragma unroll
    for (int i = 0; i < 8; ++i) {
        v[i] = s[base + tid + i * 256];
        mx = fmaxf(mx, v[i]);
    }
    red[tid] = mx;
    __syncthreads();
    for (int off = 128; off > 0; off >>= 1) {
        if (tid < off) red[tid] = fmaxf(red[tid], red[tid + off]);
        __syncthreads();
    }
    const float rowmax = red[0];
    __syncthreads();
    float sum = 0.0f;
#pragma unroll
    for (int i = 0; i < 8; ++i) {
        v[i] = __expf(v[i] - rowmax);
        sum += v[i];
    }
    red[tid] = sum;
    __syncthreads();
    for (int off = 128; off > 0; off >>= 1) {
        if (tid < off) red[tid] += red[tid + off];
        __syncthreads();
    }
    const float inv = 1.0f / red[0];
#pragma unroll
    for (int i = 0; i < 8; ++i) {
        const size_t c = base + tid + i * 256;
        const float o = v[i] * inv;
        s[c]   = o;
        p16[c] = (_Float16)o;
    }
}

// ---------------------------------------------------------------------------
// f32 -> f16 elementwise cast (weights)
// ---------------------------------------------------------------------------
__global__ void __launch_bounds__(256)
cast_f32_f16_kernel(const float* __restrict__ in, _Float16* __restrict__ out, size_t n) {
    const size_t i = (size_t)blockIdx.x * 256 + threadIdx.x;
    if (i < n) out[i] = (_Float16)in[i];
}

// ---------------------------------------------------------------------------
// Split V out of qkv (f16) into transposed per-head Vt[b,h,d,l]
// ---------------------------------------------------------------------------
__global__ void __launch_bounds__(256)
split_v_kernel(const _Float16* __restrict__ qkv, _Float16* __restrict__ vt,
               int Hc, int HDc, int Lc, int D, size_t n) {
    const size_t i = (size_t)blockIdx.x * 256 + threadIdx.x;
    if (i >= n) return;
    const int l = (int)(i % Lc);
    size_t t = i / Lc;
    const int d = (int)(t % HDc); t /= HDc;
    const int h = (int)(t % Hc);
    const int b = (int)(t / Hc);
    vt[i] = qkv[((size_t)(b * Lc + l)) * (3 * D) + 2 * D + h * HDc + d];
}

// ---------------------------------------------------------------------------
// Host orchestration
// ---------------------------------------------------------------------------
extern "C" void kernel_launch(void* const* d_in, const int* in_sizes, int n_in,
                              void* d_out, int out_size, void* d_ws, size_t ws_size,
                              hipStream_t stream) {
    (void)in_sizes; (void)n_in; (void)out_size; (void)ws_size;
    const int B = 2, L = 2048, D = 2048, H = 16, HD = 128;
    const int BL = B * L;

    const float* x      = (const float*)d_in[0];
    const float* w_qkv  = (const float*)d_in[1];
    const float* w_out  = (const float*)d_in[2];
    const float* w_in   = (const float*)d_in[3];
    const float* w_hid  = (const float*)d_in[4];
    const float* w_gate = (const float*)d_in[5];
    const float* g_attn = (const float*)d_in[6];
    const float* g_ffn  = (const float*)d_in[7];

    float* out_f  = (float*)d_out;                       // [B,L,D] (also after_attn scratch)
    float* scores = out_f + (size_t)BL * D;              // [B,H,L,L]

    // Workspace carve-out (all f16)
    _Float16* w = (_Float16*)d_ws;
    size_t off = 0;
    auto alloc = [&](size_t nelem) { _Float16* p = w + off; off += nelem; return p; };
    _Float16* wqkv_h  = alloc((size_t)3 * D * D);
    _Float16* wout_h  = alloc((size_t)D * D);
    _Float16* win_h   = alloc((size_t)D * D);
    _Float16* whid_h  = alloc((size_t)D * D);
    _Float16* wgate_h = alloc((size_t)D * D);
    _Float16* xn_h    = alloc((size_t)BL * D);
    _Float16* qkv_h   = alloc((size_t)BL * 3 * D);
    _Float16* vt_h    = alloc((size_t)B * H * HD * L);
    _Float16* attn_h  = alloc((size_t)BL * D);
    _Float16* yn_h    = alloc((size_t)BL * D);
    _Float16* hid_h   = alloc((size_t)BL * D);
    _Float16* hg_h    = alloc((size_t)BL * D);
    _Float16* p16     = alloc((size_t)B * H * L * L);    // f16 softmax probs

    const dim3 blk(256);
    auto cgrid = [](size_t n) { return dim3((unsigned)((n + 255) / 256)); };

    // 1. Cast weights to f16
    cast_f32_f16_kernel<<<cgrid((size_t)3 * D * D), blk, 0, stream>>>(w_qkv,  wqkv_h,  (size_t)3 * D * D);
    cast_f32_f16_kernel<<<cgrid((size_t)D * D),     blk, 0, stream>>>(w_out,  wout_h,  (size_t)D * D);
    cast_f32_f16_kernel<<<cgrid((size_t)D * D),     blk, 0, stream>>>(w_in,   win_h,   (size_t)D * D);
    cast_f32_f16_kernel<<<cgrid((size_t)D * D),     blk, 0, stream>>>(w_hid,  whid_h,  (size_t)D * D);
    cast_f32_f16_kernel<<<cgrid((size_t)D * D),     blk, 0, stream>>>(w_gate, wgate_h, (size_t)D * D);

    // 2. xn = rmsnorm(x) -> f16
    rmsnorm_cast_kernel<<<dim3(BL), blk, 0, stream>>>(x, g_attn, xn_h, D);

    // 3. qkv = xn @ w_qkv^T -> f16   [BL, 3D]
    wmma_gemm_nt<1><<<dim3(3 * D / 128, BL / 128, 1), blk, 0, stream>>>(
        xn_h, wqkv_h, nullptr, qkv_h, nullptr, nullptr,
        BL, 3 * D, D, D, D, 3 * D, 0, 0, 0, 0, 0, 0, 1, 1.0f);

    // 4. Vt[b,h,d,l]
    split_v_kernel<<<cgrid((size_t)B * H * HD * L), blk, 0, stream>>>(
        qkv_h, vt_h, H, HD, L, D, (size_t)B * H * HD * L);

    // 5. scores = (Q @ K^T) / sqrt(HD)   batched over (b,h): M=N=L, K=HD
    wmma_gemm_nt<0><<<dim3(L / 128, L / 128, B * H), blk, 0, stream>>>(
        qkv_h, qkv_h + D, scores, nullptr, nullptr, nullptr,
        L, L, HD, 3 * D, 3 * D, L,
        (long long)L * 3 * D, HD, (long long)L * 3 * D, HD,
        (long long)H * L * L, (long long)L * L, H, 0.08838834764831845f);

    // 6. softmax rows in place (these ARE the returned scores) + f16 probs
    softmax_rows_kernel<<<dim3(B * H * L), blk, 0, stream>>>(scores, p16, L);

    // 7. attn = P @ Vt^T -> f16 merged heads [B,L,D]
    wmma_gemm_nt<1><<<dim3(HD / 128, L / 128, B * H), blk, 0, stream>>>(
        p16, vt_h, nullptr, attn_h, nullptr, nullptr,
        L, HD, L, L, L, D,
        (long long)H * L * L, (long long)L * L,
        (long long)H * HD * L, (long long)HD * L,
        (long long)L * D, HD, H, 1.0f);

    // 8. after_attn = x + attn @ w_out^T  -> d_out (f32)
    wmma_gemm_nt<2><<<dim3(D / 128, BL / 128, 1), blk, 0, stream>>>(
        attn_h, wout_h, out_f, nullptr, x, nullptr,
        BL, D, D, D, D, D, 0, 0, 0, 0, 0, 0, 1, 1.0f);

    // 9. yn = rmsnorm(after_attn) -> f16
    rmsnorm_cast_kernel<<<dim3(BL), blk, 0, stream>>>(out_f, g_ffn, yn_h, D);

    // 10. hidden = relu(yn @ w_in^T) -> f16
    wmma_gemm_nt<3><<<dim3(D / 128, BL / 128, 1), blk, 0, stream>>>(
        yn_h, win_h, nullptr, hid_h, nullptr, nullptr,
        BL, D, D, D, D, D, 0, 0, 0, 0, 0, 0, 1, 1.0f);

    // 11. hg = hidden * (hidden @ w_gate^T) -> f16
    wmma_gemm_nt<4><<<dim3(D / 128, BL / 128, 1), blk, 0, stream>>>(
        hid_h, wgate_h, nullptr, hg_h, nullptr, hid_h,
        BL, D, D, D, D, D, 0, 0, 0, 0, 0, 0, 1, 1.0f);

    // 12. out = after_attn + hg @ w_hidden^T   (in-place residual on d_out)
    wmma_gemm_nt<2><<<dim3(D / 128, BL / 128, 1), blk, 0, stream>>>(
        hg_h, whid_h, out_f, nullptr, out_f, nullptr,
        BL, D, D, D, D, D, 0, 0, 0, 0, 0, 0, 1, 1.0f);
}
